// LSTM_27719718928936
// MI455X (gfx1250) — matrix operand order repaired and checked by reading, not compiled, not run on hardware
//
#include <hip/hip_runtime.h>
#include <hip/hip_bf16.h>

// ---------------- problem constants ----------------
#define B_   32
#define T_   2048
#define H_   256
#define L_   2
#define K_   512          // 2*H (concat [x|h])
#define NWG  16           // one WG per WGP; each owns 16 columns of H
#define HL   16           // H columns per WG
#define NLOC 64           // 4 gates * HL outputs per WG
#define NTHREADS 256      // 8 wave32

typedef __attribute__((ext_vector_type(16))) __bf16 v16bf;
typedef __attribute__((ext_vector_type(8)))  float  v8f;

// ---------------- workspace layout (bytes) ----------------
#define WT_ELEMS   (L_ * 4 * H_ * K_)          // 1,048,576 bf16 (2 MB)
#define HBUF_ELEMS (4 * B_ * H_)               // (l,parity) x B x H bf16
#define HBUF_OFF   ((size_t)WT_ELEMS * 2u)
#define CTR_OFF    (HBUF_OFF + (size_t)HBUF_ELEMS * 2u)

// LDS byte offsets (dynamic LDS base == 0: kernel has no static LDS)
#define LDSB_W     0u
#define LDSB_COMB  (L_ * NLOC * K_ * 2u)       // 131072
#define LDSB_GATES (LDSB_COMB + B_ * K_ * 2u)  // +32768
#define LDSB_C     (LDSB_GATES + 4u * B_ * HL * 4u)

__device__ __forceinline__ unsigned short f2bf(float f) {
  unsigned u = __builtin_bit_cast(unsigned, f);
  unsigned r = u + 0x7FFFu + ((u >> 16) & 1u);   // round-to-nearest-even
  return (unsigned short)(r >> 16);
}

// ---------------- prep: W fp32 -> bf16, WG-sliced layout ----------------
__global__ void lstm_prep_w(const float* __restrict__ W,
                            unsigned short* __restrict__ wt) {
  for (int d = blockIdx.x * blockDim.x + threadIdx.x; d < WT_ELEMS;
       d += gridDim.x * blockDim.x) {
    int k  = d & (K_ - 1);
    int r  = d >> 9;
    int hl = r & 15; r >>= 4;
    int g  = r & 3;  r >>= 2;
    int wg = r & 15;
    int l  = r >> 4;
    wt[d] = f2bf(W[(((size_t)l * 4 + g) * H_ + wg * HL + hl) * K_ + k]);
  }
}

__global__ void lstm_init(unsigned short* __restrict__ hbuf,
                          unsigned* __restrict__ ctr) {
  for (int i = threadIdx.x; i < HBUF_ELEMS; i += blockDim.x) hbuf[i] = 0;
  if (threadIdx.x == 0) *ctr = 0u;
}

// ---------------- device-wide barrier (16 resident WGs) ----------------
__device__ __forceinline__ void grid_barrier(unsigned* ctr, unsigned target) {
  __threadfence();                       // publish h stores at device scope
  __syncthreads();
  __builtin_amdgcn_s_cluster_barrier();  // HW fast path if cluster-dispatched; NOP otherwise
  if (threadIdx.x == 0) {
    __hip_atomic_fetch_add(ctr, 1u, __ATOMIC_RELEASE, __HIP_MEMORY_SCOPE_AGENT);
    while (__hip_atomic_load(ctr, __ATOMIC_ACQUIRE, __HIP_MEMORY_SCOPE_AGENT) < target)
      __builtin_amdgcn_s_sleep(2);
  }
  __syncthreads();
}

// ---------------- TDM: DMA 32x(512B) h-rows from global into LDS with ----------
// ---------------- 512B padding after each row (interleaves [x | h])  ----------
// D# per CDNA5 ISA ch.8: group0 {count, lds_addr, global_addr, type=2},
// group1 {data_size=4B, pad_enable, pad_interval=512B, pad_amount=512B,
//         tensor_dim0=128 dw, tensor_dim1=32, tile_dim0=128 dw, tile_dim1=32,
//         dim0_stride=128 dw (contiguous source)}. Groups 2/3 unused (2-D).
__device__ __forceinline__ void tdm_load_h_rows(const void* gsrc,
                                                unsigned lds_byte_off) {
  typedef unsigned int u32x4 __attribute__((ext_vector_type(4)));
  typedef int          i32x8 __attribute__((ext_vector_type(8)));
  typedef int          i32x4 __attribute__((ext_vector_type(4)));
  const unsigned long long ga = (unsigned long long)gsrc;
  u32x4 g0;
  g0[0] = 1u;                                            // count=1, user D#
  g0[1] = lds_byte_off;                                  // lds_addr
  g0[2] = (unsigned)(ga & 0xFFFFFFFFu);                  // global_addr[31:0]
  g0[3] = (unsigned)((ga >> 32) & 0x01FFFFFFu) | (2u << 30);  // addr[56:32]|type=2
  i32x8 g1;
  g1[0] = (int)((127u << 25) | (6u << 22) | (1u << 20) | (2u << 16));
  //            pad_amount=128dw  pad_int=128dw  pad_en    data_size=4B
  g1[1] = (int)(128u << 16);   // tensor_dim0[15:0]=128 (dw1[31:16])
  g1[2] = (int)(32u << 16);    // tensor_dim0 hi=0 | tensor_dim1=32
  g1[3] = (int)(128u << 16);   // tensor_dim1 hi=0 | tile_dim0=128
  g1[4] = 32;                  // tile_dim1=32, tile_dim2=0
  g1[5] = 128;                 // tensor_dim0_stride=128 dw (contiguous)
  g1[6] = 0;
  g1[7] = 0;
  const i32x4 z4 = {0, 0, 0, 0};
  const i32x8 z8 = {0, 0, 0, 0, 0, 0, 0, 0};
  __builtin_amdgcn_tensor_load_to_lds(g0, g1, z4, z4, z8, 0);  // TENSORcnt++
}

// ---------------- one LSTM layer-step for this WG ----------------
__device__ __forceinline__ void lstm_layer(
    int l, int t, int wg, int tid,
    const unsigned short* __restrict__ lds_w,
    const unsigned short* __restrict__ lds_comb,
    float* __restrict__ lds_gates, float* __restrict__ lds_c,
    const float* __restrict__ bias,
    unsigned short* __restrict__ hbuf_out,      // hbuf[l][t&1]
    float* __restrict__ out)                    // non-null for layer 1
{
  const int lane  = tid & 31;
  const int wv    = tid >> 5;
  const int mtile = wv & 1;       // batch 16-row tile
  const int g     = wv >> 1;      // gate 0..3  (N 16-col tile)
  const int lrow  = lane & 15;
  const int khalf = lane >> 4;

  // bias folded into accumulator init (same N column -> all 8 M rows)
  const float bv = bias[(l * 4 + g) * H_ + wg * HL + lrow];
  v8f acc;
#pragma unroll
  for (int i = 0; i < 8; ++i) acc[i] = bv;

  const unsigned short* arow = lds_comb + (mtile * 16 + lrow) * K_ + khalf * 8;
  const unsigned short* brow = lds_w + (l * NLOC + g * HL + lrow) * K_ + khalf * 8;

#pragma unroll
  for (int kb = 0; kb < K_ / 32; ++kb) {
    union { v16bf v; uint4 u[2]; } A, Bm;
    A.u[0]  = *(const uint4*)(arow + kb * 32);
    A.u[1]  = *(const uint4*)(arow + kb * 32 + 16);
    Bm.u[0] = *(const uint4*)(brow + kb * 32);
    Bm.u[1] = *(const uint4*)(brow + kb * 32 + 16);
    acc = __builtin_amdgcn_wmma_f32_16x16x32_bf16(
        false, A.v, false, Bm.v, (short)0, acc, false, false);
  }

#pragma unroll
  for (int j = 0; j < 8; ++j) {
    int brow_i = mtile * 16 + khalf * 8 + j;
    lds_gates[(g * B_ + brow_i) * HL + lrow] = acc[j];
  }
  __syncthreads();

#pragma unroll
  for (int r = 0; r < 2; ++r) {
    int e  = tid + r * NTHREADS;
    int b  = e >> 4, hl = e & 15;
    float gi = lds_gates[(0 * B_ + b) * HL + hl];
    float gf = lds_gates[(1 * B_ + b) * HL + hl];
    float go = lds_gates[(2 * B_ + b) * HL + hl];
    float gc = lds_gates[(3 * B_ + b) * HL + hl];
    float ig = 1.f / (1.f + __expf(-gi));
    float fg = 1.f / (1.f + __expf(-gf));
    float og = 1.f / (1.f + __expf(-go));
    float ch = tanhf(gc);
    float cn = fg * lds_c[(l * B_ + b) * HL + hl] + ig * ch;
    lds_c[(l * B_ + b) * HL + hl] = cn;
    float hn = og * tanhf(cn);
    hbuf_out[b * H_ + wg * HL + hl] = f2bf(hn);
    if (out) {
      out[((size_t)b * T_ + t) * H_ + wg * HL + hl] = hn;   // outputs[B,T,H]
      if (t == T_ - 1) {
        out[(size_t)B_ * T_ * H_ + b * H_ + wg * HL + hl] = hn;            // hT[-1]
        out[(size_t)B_ * T_ * H_ + B_ * H_ + b * H_ + wg * HL + hl] = cn;  // cT[-1]
      }
    }
  }
}

// ---------------- persistent cooperative kernel (16 WGs = 16 WGPs) ----------------
extern "C" __global__ void __launch_bounds__(NTHREADS)
lstm_persistent(const float* __restrict__ x, const float* __restrict__ bias,
                const unsigned short* __restrict__ wt_g,
                unsigned short* __restrict__ hbuf, unsigned* __restrict__ ctr,
                float* __restrict__ out) {
  extern __shared__ unsigned short smem[];
  unsigned short* lds_w    = smem;                       // 128 KB, pinned weights
  unsigned short* lds_comb = smem + L_ * NLOC * K_;      //  32 KB [32][512] bf16
  float* lds_gates = (float*)(lds_comb + B_ * K_);       //   8 KB
  float* lds_c     = lds_gates + 4 * B_ * HL;            //   4 KB

  const int tid = threadIdx.x;
  const int wg  = blockIdx.x;
  const int wv  = tid >> 5;

  // pin this WG's weight slice (both layers) into LDS for all 2048 steps
  for (int l = 0; l < L_; ++l) {
    const uint4* s = (const uint4*)(wt_g + (size_t)((l * NWG + wg) * NLOC) * K_);
    uint4*       d = (uint4*)(lds_w + l * NLOC * K_);
    for (int i = tid; i < NLOC * K_ / 8; i += NTHREADS) d[i] = s[i];
  }
  for (int i = tid; i < L_ * B_ * HL; i += NTHREADS) lds_c[i] = 0.f;
  __syncthreads();

  unsigned epoch = 0;
  for (int t = 0; t < T_; ++t) {
    const int par = t & 1, pprev = par ^ 1;

    // ---------- layer 0: comb = [bf16(x_t) | h0(t-1)] ----------
    // TDM DMAs h0(t-1) into the second half of every comb row (pad interleave)
    if (wv == 0)
      tdm_load_h_rows(hbuf + (0 * 2 + pprev) * B_ * H_, LDSB_COMB + H_ * 2u);
    // meanwhile all waves convert x_t fp32->bf16 into the first halves
    for (int i = tid; i < B_ * H_ / 4; i += NTHREADS) {
      int e = i * 4, b = e >> 8, k = e & 255;
      const float4 v = *(const float4*)(x + ((size_t)b * T_ + t) * H_ + k);
      ushort4 p;
      p.x = f2bf(v.x); p.y = f2bf(v.y); p.z = f2bf(v.z); p.w = f2bf(v.w);
      *(ushort4*)(lds_comb + b * K_ + k) = p;
      if (t + 1 < T_)
        __builtin_prefetch(x + ((size_t)b * T_ + t + 1) * H_ + k, 0, 1);
    }
    if (wv == 0) __builtin_amdgcn_s_wait_tensorcnt(0);
    __syncthreads();
    lstm_layer(0, t, wg, tid, lds_w, lds_comb, lds_gates, lds_c, bias,
               hbuf + (0 * 2 + par) * B_ * H_, nullptr);
    grid_barrier(ctr, (++epoch) * NWG);      // publish h0(t) to all WGPs

    // ---------- layer 1: comb = [h0(t) | h1(t-1)] ----------
    if (wv == 0) {
      tdm_load_h_rows(hbuf + (0 * 2 + par)   * B_ * H_, LDSB_COMB);            // x-half
      tdm_load_h_rows(hbuf + (1 * 2 + pprev) * B_ * H_, LDSB_COMB + H_ * 2u);  // h-half
      __builtin_amdgcn_s_wait_tensorcnt(0);
    }
    __syncthreads();
    lstm_layer(1, t, wg, tid, lds_w, lds_comb, lds_gates, lds_c, bias,
               hbuf + (1 * 2 + par) * B_ * H_, out);
    grid_barrier(ctr, (++epoch) * NWG);      // publish h1(t); WAR-safe via parity buffers
  }
}

// ---------------- host entry ----------------
extern "C" void kernel_launch(void* const* d_in, const int* in_sizes, int n_in,
                              void* d_out, int out_size, void* d_ws, size_t ws_size,
                              hipStream_t stream) {
  const float* x    = (const float*)d_in[0];   // [B,T,H] fp32
  const float* W    = (const float*)d_in[1];   // [L,4,H,2H] fp32
  const float* bias = (const float*)d_in[2];   // [L,4,H] fp32

  unsigned short* wt   = (unsigned short*)d_ws;
  unsigned short* hbuf = (unsigned short*)((char*)d_ws + HBUF_OFF);
  unsigned*       ctr  = (unsigned*)((char*)d_ws + CTR_OFF);
  float*          out  = (float*)d_out;

  lstm_prep_w<<<512, 256, 0, stream>>>(W, wt);
  lstm_init<<<1, 256, 0, stream>>>(hbuf, ctr);

  const size_t lds_bytes =
      (size_t)(L_ * NLOC * K_ + B_ * K_) * 2 + (size_t)(4 * B_ * HL + L_ * B_ * HL) * 4;
  lstm_persistent<<<NWG, NTHREADS, lds_bytes, stream>>>(x, bias, wt, hbuf, ctr, out);
}
